// SlotAttentionAutoEncoder_61108794687666
// MI455X (gfx1250) — matrix operand order repaired and checked by reading, not compiled, not run on hardware
//
#include <hip/hip_runtime.h>
#include <hip/hip_bf16.h>
#include <math.h>

// ---------------------------------------------------------------------------
// SlotAttention autoencoder for MI455X (gfx1250).
// Heavy 64-channel 5x5 convs + 64x64 linears -> v_wmma_f32_16x16x32_f16
// (f16 operands, f32 accumulate), with weights pre-swizzled into the exact
// WMMA B-fragment register layout so each fragment is a single 32B load,
// and A fragments read from LDS as 2x b128.  Two output rows per wave to
// reuse each B fragment across two WMMAs.
// Dilated deconvs / tiny slot ops -> VALU.  Requires ws_size >= ~345 MB.
// ---------------------------------------------------------------------------

typedef __attribute__((ext_vector_type(16))) _Float16 v16h;
typedef __attribute__((ext_vector_type(8)))  _Float16 v8h;
typedef __attribute__((ext_vector_type(4)))  _Float16 v4h;
typedef __attribute__((ext_vector_type(8)))  float    v8f;
typedef __attribute__((ext_vector_type(4)))  float    v4f;

union U16h { v16h v; v8h h[2]; _Float16 e[16]; };
union U8f  { v8f  v; float    e[8];  };

#define HH 112
#define WW 112
#define HWP 12544          // 112*112
#define BB 8
#define DD 64
#define NSL 7
#define BN 56              // B*NS
#define ROWS 100352        // B*HW

__device__ inline v4h load4h(const float* p) {
  const v4f x = *(const v4f*)p;
  v4h h;
  h.x = (_Float16)x.x; h.y = (_Float16)x.y;
  h.z = (_Float16)x.z; h.w = (_Float16)x.w;
  return h;
}
__device__ inline v4h load4h(const _Float16* p) { return *(const v4h*)p; }

// ---------------------------------------------------------------------------
// WMMA 5x5 conv, Cin=Cout=64, pad=2, stride 1, NHWC.
// One wave32 per block: 2 rows x 16 pixels x 64 cout.
// wfrag layout: frag f = ((kh*5+kw)*2+kc)*4+nt ; halfs [f][lane][16] contiguous.
// ---------------------------------------------------------------------------
template<typename TIN, typename TOUT, bool RELU>
__global__ __launch_bounds__(32)
void conv5x5_64_wmma(const TIN* __restrict__ in, const _Float16* __restrict__ wfrag,
                     const float* __restrict__ bias, TOUT* __restrict__ out,
                     int N, int H, int W) {
  const int lane = threadIdx.x;
  const int w0 = blockIdx.x * 16;
  const int h0 = blockIdx.y * 2;
  const int n  = blockIdx.z;
  const int lhalf = lane >> 4;     // 0|1
  const int lmod  = lane & 15;

  __shared__ __align__(16) _Float16 tile[2][20 * 64];  // 2 rows x 20 px x 64 cin

  U8f acc[2][4];
  for (int m = 0; m < 2; ++m)
    for (int t = 0; t < 4; ++t)
      acc[m][t].v = (v8f){0.f,0.f,0.f,0.f,0.f,0.f,0.f,0.f};

  for (int kh = 0; kh < 5; ++kh) {
    const int ihA = h0 + kh - 2;                       // input row for out row h0
    const bool vld[2] = { (ihA >= 0) && (ihA < H),
                          (ihA + 1 >= 0) && (ihA + 1 < H) };
    for (int m = 0; m < 2; ++m) {
      if (!vld[m]) continue;
      const TIN* rowp = in + ((size_t)(n * H + (ihA + m))) * W * 64;
      for (int i = lane; i < 20 * 16; i += 32) {
        int p = i >> 4, c4 = (i & 15) << 2;
        int iw = w0 - 2 + p;
        v4h hv;
        if (iw >= 0 && iw < W) hv = load4h(rowp + (size_t)iw * 64 + c4);
        else { hv.x = hv.y = hv.z = hv.w = (_Float16)0.f; }
        *(v4h*)&tile[m][p * 64 + c4] = hv;
      }
    }
    __syncthreads();

    for (int kw = 0; kw < 5; ++kw) {
      const int pxoff = (lmod + kw) * 64;
      for (int kc = 0; kc < 2; ++kc) {
        // A fragments (ISA 16-bit A 16x32 layout): two contiguous 8-half runs
        U16h A[2] = {};
        const int koff = kc * 32 + lhalf * 8;
        for (int m = 0; m < 2; ++m) {
          if (!vld[m]) continue;
          const _Float16* tp = &tile[m][pxoff + koff];
          A[m].h[0] = *(const v8h*)tp;          // K = base .. base+7
          A[m].h[1] = *(const v8h*)(tp + 16);   // K = base+16 .. base+23
        }
        const int fbase = ((kh * 5 + kw) * 2 + kc) * 4;
        for (int nt = 0; nt < 4; ++nt) {
          U16h Bf;
          Bf.v = *(const v16h*)(wfrag + ((size_t)(fbase + nt)) * 512 + lane * 16);
          if (vld[0])
            acc[0][nt].v = __builtin_amdgcn_wmma_f32_16x16x32_f16(
                false, A[0].v, false, Bf.v, (short)0, acc[0][nt].v, false, false);
          if (vld[1])
            acc[1][nt].v = __builtin_amdgcn_wmma_f32_16x16x32_f16(
                false, A[1].v, false, Bf.v, (short)0, acc[1][nt].v, false, false);
        }
      }
    }
    __syncthreads();
  }

  for (int m = 0; m < 2; ++m) {
    TOUT* orow = out + ((size_t)(n * H + (h0 + m))) * W * 64;
    for (int nt = 0; nt < 4; ++nt) {
      const int cout = nt * 16 + lmod;
      const float bv = bias[cout];
      for (int r = 0; r < 8; ++r) {
        int px = w0 + r + lhalf * 8;           // C/D layout: M = r + 8*(lane/16)
        float vv = acc[m][nt].e[r] + bv;
        if (RELU) vv = vv > 0.f ? vv : 0.f;
        orow[(size_t)px * 64 + cout] = (TOUT)vv;
      }
    }
  }
}

// ---------------------------------------------------------------------------
// WMMA linear: Y = [relu](X @ W^T + b), X (R,64) f32.
// wfrag: frag f = kc*4+nt ; halfs [f][lane][16] contiguous (B layout).
// ---------------------------------------------------------------------------
template<bool RELU>
__global__ __launch_bounds__(32)
void linear64_wmma(const float* __restrict__ X, const _Float16* __restrict__ wfrag,
                   const float* __restrict__ bias, float* __restrict__ Y, int R) {
  const int lane = threadIdx.x;
  const int r0 = blockIdx.x * 16;
  const int lhalf = lane >> 4, lmod = lane & 15;
  __shared__ __align__(16) _Float16 tile[16 * 64];
  for (int i = lane; i < 16 * 16; i += 32) {
    int rr = i >> 4, c4 = (i & 15) << 2;
    *(v4h*)&tile[rr * 64 + c4] = load4h(X + (size_t)(r0 + rr) * 64 + c4);
  }
  __syncthreads();

  U8f acc[4];
  for (int t = 0; t < 4; ++t) acc[t].v = (v8f){0.f,0.f,0.f,0.f,0.f,0.f,0.f,0.f};

  for (int kc = 0; kc < 2; ++kc) {
    U16h A;
    const _Float16* tp = &tile[lmod * 64 + kc * 32 + lhalf * 8];
    A.h[0] = *(const v8h*)tp;
    A.h[1] = *(const v8h*)(tp + 16);
    for (int nt = 0; nt < 4; ++nt) {
      U16h Bf;
      Bf.v = *(const v16h*)(wfrag + ((size_t)(kc * 4 + nt)) * 512 + lane * 16);
      acc[nt].v = __builtin_amdgcn_wmma_f32_16x16x32_f16(
          false, A.v, false, Bf.v, (short)0, acc[nt].v, false, false);
    }
  }
  for (int nt = 0; nt < 4; ++nt) {
    const int cout = nt * 16 + lmod;
    const float bv = bias[cout];
    for (int r = 0; r < 8; ++r) {
      int row = r0 + r + lhalf * 8;
      float vv = acc[nt].e[r] + bv;
      if (RELU) vv = vv > 0.f ? vv : 0.f;
      Y[(size_t)row * 64 + cout] = vv;
    }
  }
}

// ---------------------------------------------------------------------------
// Small helpers / direct kernels
// ---------------------------------------------------------------------------
__global__ void zero_f(float* p, int n) {
  int i = blockIdx.x * 256 + threadIdx.x;
  if (i < n) p[i] = 0.f;
}
__global__ void copy_f(const float* s, float* d, int n) {
  int i = blockIdx.x * 256 + threadIdx.x;
  if (i < n) d[i] = s[i];
}
__global__ void add_f(float* a, const float* b, int n) {
  int i = blockIdx.x * 256 + threadIdx.x;
  if (i < n) a[i] += b[i];
}

// enc conv1: image NCHW (8,3,112,112) -> NHWC (8,112,112,64), relu
__global__ void conv1_direct(const float* __restrict__ img, const float* __restrict__ w,
                             const float* __restrict__ b, float* __restrict__ out) {
  int idx = blockIdx.x * 256 + threadIdx.x;
  if (idx >= BB * HWP * DD) return;
  int o = idx & 63;
  int t = idx >> 6;
  int x = t % WW; t /= WW;
  int y = t % HH; int n = t / HH;
  float acc = b[o];
  for (int c = 0; c < 3; ++c)
    for (int kh = 0; kh < 5; ++kh) {
      int ih = y + kh - 2; if (ih < 0 || ih >= HH) continue;
      for (int kw = 0; kw < 5; ++kw) {
        int iw = x + kw - 2; if (iw < 0 || iw >= WW) continue;
        acc += img[((n * 3 + c) * HH + ih) * WW + iw] * w[((o * 3 + c) * 5 + kh) * 5 + kw];
      }
    }
  out[idx] = acc > 0.f ? acc : 0.f;
}

// add encoder positional embedding in place (NHWC)
__global__ void pos_add_enc(float* x, const float* __restrict__ pw, const float* __restrict__ pb) {
  int idx = blockIdx.x * 256 + threadIdx.x;
  if (idx >= BB * HWP * DD) return;
  int d = idx & 63;
  int t = idx >> 6;
  int wq = t % WW; t /= WW;
  int hq = t % HH;
  float gx = wq * (1.f / 111.f), gy = hq * (1.f / 111.f);
  float pe = gx * pw[d * 4 + 0] + gy * pw[d * 4 + 1] +
             (1.f - gx) * pw[d * 4 + 2] + (1.f - gy) * pw[d * 4 + 3] + pb[d];
  x[idx] += pe;
}

// per-batch mean/var partial sums (802816 elems per batch)
__global__ void reduce_stats(const float* __restrict__ x, float* stats) {
  int b = blockIdx.y;
  int tid = threadIdx.x;
  const float* xb = x + (size_t)b * 802816;
  int base = blockIdx.x * 1024 + tid;
  float s = 0.f, s2 = 0.f;
  for (int k = 0; k < 4; ++k) {
    float v = xb[base + k * 256];
    s += v; s2 += v * v;
  }
  __shared__ float r1[256], r2[256];
  r1[tid] = s; r2[tid] = s2; __syncthreads();
  for (int st = 128; st > 0; st >>= 1) {
    if (tid < st) { r1[tid] += r1[tid + st]; r2[tid] += r2[tid + st]; }
    __syncthreads();
  }
  if (tid == 0) { atomicAdd(&stats[b * 2], r1[0]); atomicAdd(&stats[b * 2 + 1], r2[0]); }
}
__global__ void apply_gln(float* x, const float* __restrict__ stats) {
  int idx = blockIdx.x * 256 + threadIdx.x;
  if (idx >= BB * HWP * DD) return;
  int b = idx / 802816;
  float mu = stats[b * 2] * (1.f / 802816.f);
  float var = stats[b * 2 + 1] * (1.f / 802816.f) - mu * mu;
  x[idx] = (x[idx] - mu) * rsqrtf(var + 1e-5f);
}

// per-row layernorm (D=64) with gamma/beta. block = 64 threads, grid = rows
__global__ void ln_rows(const float* __restrict__ X, const float* __restrict__ g,
                        const float* __restrict__ bb, float* __restrict__ Y, int R) {
  int r = blockIdx.x;
  int d = threadIdx.x;
  __shared__ float s1[64], s2[64];
  float x = X[(size_t)r * 64 + d];
  s1[d] = x; s2[d] = x * x; __syncthreads();
  for (int st = 32; st > 0; st >>= 1) {
    if (d < st) { s1[d] += s1[d + st]; s2[d] += s2[d + st]; }
    __syncthreads();
  }
  float mu = s1[0] * (1.f / 64.f);
  float var = s2[0] * (1.f / 64.f) - mu * mu;
  Y[(size_t)r * 64 + d] = (x - mu) * rsqrtf(var + 1e-5f) * g[d] + bb[d];
}

// attn: per token j, 7 dots, softmax over slots, +EPS, store; accumulate S[b,i]
__global__ void attn_dots(const float* __restrict__ q, const float* __restrict__ k,
                          float* __restrict__ attn7, float* S) {
  __shared__ float qs[NSL * 64];
  __shared__ float red[256];
  int b = blockIdx.y;
  int tid = threadIdx.x;
  for (int i = tid; i < NSL * 64; i += 256) qs[i] = q[b * NSL * 64 + i];
  __syncthreads();
  int j = blockIdx.x * 256 + tid;
  float a[NSL];
  if (j < HWP) {
    float dot[NSL];
    for (int i = 0; i < NSL; ++i) dot[i] = 0.f;
    const float* kv = k + ((size_t)b * HWP + j) * 64;
    for (int d = 0; d < 64; ++d) {
      float kd = kv[d];
      for (int i = 0; i < NSL; ++i) dot[i] += qs[i * 64 + d] * kd;
    }
    float mx = dot[0] * 0.125f;
    for (int i = 1; i < NSL; ++i) { float v = dot[i] * 0.125f; mx = v > mx ? v : mx; }
    float s = 0.f;
    for (int i = 0; i < NSL; ++i) { a[i] = expf(dot[i] * 0.125f - mx); s += a[i]; }
    float inv = 1.f / s;
    float* ap = attn7 + ((size_t)b * HWP + j) * 8;
    for (int i = 0; i < NSL; ++i) { a[i] = a[i] * inv + 1e-8f; ap[i] = a[i]; }
  } else {
    for (int i = 0; i < NSL; ++i) a[i] = 0.f;
  }
  for (int i = 0; i < NSL; ++i) {
    red[tid] = a[i]; __syncthreads();
    for (int st = 128; st > 0; st >>= 1) {
      if (tid < st) red[tid] += red[tid + st];
      __syncthreads();
    }
    if (tid == 0) atomicAdd(&S[b * NSL + i], red[0]);
    __syncthreads();
  }
}

// updates[b,i,d] = sum_j attn7[b,j,i] * v[b,j,d]  (chunked + atomic)
__global__ void attn_updates(const float* __restrict__ attn7, const float* __restrict__ v,
                             float* upd) {
  int b = blockIdx.z, i = blockIdx.y, chunk = blockIdx.x;
  int t = threadIdx.x;
  int d = t & 63, g = t >> 6;
  __shared__ float red[256];
  float acc = 0.f;
  for (int jj = g; jj < 1024; jj += 4) {
    int j = chunk * 1024 + jj;
    if (j < HWP) {
      size_t base = (size_t)b * HWP + j;
      acc += attn7[base * 8 + i] * v[base * 64 + d];
    }
  }
  red[t] = acc; __syncthreads();
  if (g == 0) {
    float s = red[d] + red[64 + d] + red[128 + d] + red[192 + d];
    atomicAdd(&upd[((size_t)b * NSL + i) * 64 + d], s);
  }
}

__global__ void div_upd(float* upd, const float* __restrict__ S) {
  int idx = blockIdx.x * 256 + threadIdx.x;
  if (idx >= BN * 64) return;
  upd[idx] /= S[idx >> 6];
}

// small dense: Y[r,o] = [relu](X[r,:64] . W[o,:64] + b[o])
__global__ void linear_small(const float* __restrict__ X, const float* __restrict__ W,
                             const float* __restrict__ b, float* __restrict__ Y,
                             int R, int O, int relu) {
  int idx = blockIdx.x * 256 + threadIdx.x;
  if (idx >= R * O) return;
  int o = idx % O, r = idx / O;
  const float* xr = X + r * 64;
  const float* wr = W + o * 64;
  float acc = b[o];
  for (int c = 0; c < 64; ++c) acc += xr[c] * wr[c];
  if (relu) acc = acc > 0.f ? acc : 0.f;
  Y[idx] = acc;
}

__global__ void gru_combine(const float* __restrict__ gi, const float* __restrict__ gh,
                            float* slots) {
  int idx = blockIdx.x * 256 + threadIdx.x;
  if (idx >= BN * 64) return;
  int r = idx >> 6, d = idx & 63;
  const float* gir = gi + r * 192;
  const float* ghr = gh + r * 192;
  float ir = gir[d], iz = gir[64 + d], inn = gir[128 + d];
  float hr = ghr[d], hz = ghr[64 + d], hn = ghr[128 + d];
  float rr = 1.f / (1.f + expf(-(ir + hr)));
  float z  = 1.f / (1.f + expf(-(iz + hz)));
  float nn = tanhf(inn + rr * hn);
  float h0 = slots[idx];
  slots[idx] = (1.f - z) * nn + z * h0;
}

// decoder input: broadcast slots + pos_d -> (56,8,8,64) f16 NHWC
__global__ void build_decin(const float* __restrict__ slots, const float* __restrict__ pw,
                            const float* __restrict__ pb, _Float16* out) {
  int idx = blockIdx.x * 256 + threadIdx.x;
  if (idx >= BN * 8 * 8 * 64) return;
  int d = idx & 63;
  int t = idx >> 6;
  int pwq = t & 7; t >>= 3;
  int phq = t & 7; int n = t >> 3;
  float gx = pwq * (1.f / 7.f), gy = phq * (1.f / 7.f);
  float pe = gx * pw[d * 4 + 0] + gy * pw[d * 4 + 1] +
             (1.f - gx) * pw[d * 4 + 2] + (1.f - gy) * pw[d * 4 + 3] + pb[d];
  out[idx] = (_Float16)(slots[n * 64 + d] + pe);
}

// transposed conv (lhs_dilation=2, pad (2,3)), 5x5 64->64, gather form, NHWC f16
__global__ void deconv5x5(const _Float16* __restrict__ in, const float* __restrict__ w,
                          const float* __restrict__ b, _Float16* out, int N, int Hi, int Wi) {
  int Ho = Hi * 2, Wo = Wi * 2;
  size_t total = (size_t)N * Ho * Wo * 64;
  size_t idx = (size_t)blockIdx.x * 256 + threadIdx.x;
  if (idx >= total) return;
  int o = idx & 63;
  size_t t = idx >> 6;
  int q = t % Wo; t /= Wo;
  int p = t % Ho; int n = (int)(t / Ho);
  float acc = b[o];
  for (int kh = 0; kh < 5; ++kh) {
    int u = p + kh - 2;
    if (u < 0 || (u & 1) || (u >> 1) >= Hi) continue;
    int ih = u >> 1;
    for (int kw = 0; kw < 5; ++kw) {
      int vq = q + kw - 2;
      if (vq < 0 || (vq & 1) || (vq >> 1) >= Wi) continue;
      int iw = vq >> 1;
      const _Float16* ip = in + ((size_t)(n * Hi + ih) * Wi + iw) * 64;
      const float* wp = w + (size_t)o * 1600 + kh * 5 + kw;   // (O,C,5,5)
      for (int c = 0; c < 64; ++c) acc += (float)ip[c] * wp[c * 25];
    }
  }
  acc = acc > 0.f ? acc : 0.f;
  out[idx] = (_Float16)acc;
}

// final 3x3 conv 64->4 with crop to 112x112, split into recons/masks
__global__ void conv6_out(const _Float16* __restrict__ in, const float* __restrict__ w,
                          const float* __restrict__ b, float* recons, float* masks) {
  int idx = blockIdx.x * 256 + threadIdx.x;
  if (idx >= BN * HWP * 4) return;
  int c4 = idx & 3;
  int t = idx >> 2;
  int x = t % WW; t /= WW;
  int y = t % HH; int n = t / HH;
  float acc = b[c4];
  for (int kh = 0; kh < 3; ++kh) {
    int ih = y + kh - 1; if (ih < 0) continue;     // ih <= 112 < 128 always
    for (int kw = 0; kw < 3; ++kw) {
      int iw = x + kw - 1; if (iw < 0) continue;
      const _Float16* ip = in + ((size_t)(n * 128 + ih) * 128 + iw) * 64;
      const float* wp = w + (size_t)c4 * 576 + kh * 3 + kw;   // (4,64,3,3)
      for (int c = 0; c < 64; ++c) acc += (float)ip[c] * wp[c * 9];
    }
  }
  if (c4 < 3) recons[((size_t)n * 3 + c4) * HWP + y * WW + x] = acc;
  else        masks[(size_t)n * HWP + y * WW + x] = acc;
}

// weight converters -> WMMA B-fragment register layout
// conv: frag f = ((kh*5+kw)*2+kc)*4+nt ; dst[f*512 + lane*16 + hh]
__global__ void wcvt_conv(const float* __restrict__ src, _Float16* dst) {
  int idx = blockIdx.x * 256 + threadIdx.x;
  if (idx >= 64 * 64 * 25) return;
  int hh = idx & 15;
  int l  = (idx >> 4) & 31;
  int f  = idx >> 9;
  int nt = f & 3;
  int kc = (f >> 2) & 1;
  int kwq = (f >> 3) % 5;
  int khq = f / 40;
  int cin  = kc * 32 + (hh >> 3) * 16 + (l >> 4) * 8 + (hh & 7);
  int cout = nt * 16 + (l & 15);
  dst[idx] = (_Float16)src[((cout * 64 + cin) * 5 + khq) * 5 + kwq];  // (O,C,5,5)
}
// linear: frag f = kc*4+nt ; dst[f*512 + lane*16 + hh]
__global__ void wcvt_lin(const float* __restrict__ src, _Float16* dst) {
  int idx = blockIdx.x * 256 + threadIdx.x;
  if (idx >= 4096) return;
  int hh = idx & 15;
  int l  = (idx >> 4) & 31;
  int f  = idx >> 9;
  int nt = f & 3;
  int kc = f >> 2;
  int cin  = kc * 32 + (hh >> 3) * 16 + (l >> 4) * 8 + (hh & 7);
  int cout = nt * 16 + (l & 15);
  dst[idx] = (_Float16)src[cout * 64 + cin];                          // (O,C)
}

// ---------------------------------------------------------------------------
extern "C" void kernel_launch(void* const* d_in, const int* in_sizes, int n_in,
                              void* d_out, int out_size, void* d_ws, size_t ws_size,
                              hipStream_t stream) {
  (void)in_sizes; (void)n_in; (void)out_size; (void)ws_size;
  const float* image    = (const float*)d_in[0];
  const float* init_sl  = (const float*)d_in[1];
  const float* enc_w1   = (const float*)d_in[2];  const float* enc_b1 = (const float*)d_in[3];
  const float* enc_w2   = (const float*)d_in[4];  const float* enc_b2 = (const float*)d_in[5];
  const float* enc_w3   = (const float*)d_in[6];  const float* enc_b3 = (const float*)d_in[7];
  const float* enc_w4   = (const float*)d_in[8];  const float* enc_b4 = (const float*)d_in[9];
  const float* enc_pw   = (const float*)d_in[10]; const float* enc_pb = (const float*)d_in[11];
  const float* fc1_w    = (const float*)d_in[12]; const float* fc1_b  = (const float*)d_in[13];
  const float* fc2_w    = (const float*)d_in[14]; const float* fc2_b  = (const float*)d_in[15];
  const float* ni_g     = (const float*)d_in[16]; const float* ni_b   = (const float*)d_in[17];
  const float* ns_g     = (const float*)d_in[18]; const float* ns_b   = (const float*)d_in[19];
  const float* npf_g    = (const float*)d_in[20]; const float* npf_b  = (const float*)d_in[21];
  const float* q_w      = (const float*)d_in[22]; const float* q_b    = (const float*)d_in[23];
  const float* k_w      = (const float*)d_in[24]; const float* k_b    = (const float*)d_in[25];
  const float* v_w      = (const float*)d_in[26]; const float* v_b    = (const float*)d_in[27];
  const float* gru_wih  = (const float*)d_in[28]; const float* gru_whh= (const float*)d_in[29];
  const float* gru_bih  = (const float*)d_in[30]; const float* gru_bhh= (const float*)d_in[31];
  const float* s1_w     = (const float*)d_in[32]; const float* s1_b   = (const float*)d_in[33];
  const float* s2_w     = (const float*)d_in[34]; const float* s2_b   = (const float*)d_in[35];
  const float* dec_pw   = (const float*)d_in[36]; const float* dec_pb = (const float*)d_in[37];
  const float* dec_w1   = (const float*)d_in[38]; const float* dec_b1 = (const float*)d_in[39];
  const float* dec_w2   = (const float*)d_in[40]; const float* dec_b2 = (const float*)d_in[41];
  const float* dec_w3   = (const float*)d_in[42]; const float* dec_b3 = (const float*)d_in[43];
  const float* dec_w4   = (const float*)d_in[44]; const float* dec_b4 = (const float*)d_in[45];
  const float* dec_w5   = (const float*)d_in[46]; const float* dec_b5 = (const float*)d_in[47];
  const float* dec_w6   = (const float*)d_in[48]; const float* dec_b6 = (const float*)d_in[49];

  float* out_recons = (float*)d_out;                        // 56*3*112*112
  float* out_masks  = (float*)d_out + (size_t)BN * 3 * HWP; // 56*1*112*112
  float* out_slots  = (float*)d_out + (size_t)BN * 4 * HWP; // 8*7*64

  // ---- workspace carve-up ----
  char* ws = (char*)d_ws;
  size_t off = 0;
  auto arena = [&](size_t bytes) -> char* {
    char* p = ws + off;
    off += (bytes + 255) & ~(size_t)255;
    return p;
  };
  _Float16* W2_16  = (_Float16*)arena(102400 * 2);
  _Float16* W3_16  = (_Float16*)arena(102400 * 2);
  _Float16* W4_16  = (_Float16*)arena(102400 * 2);
  _Float16* W5_16  = (_Float16*)arena(102400 * 2);
  _Float16* FC1_16 = (_Float16*)arena(4096 * 2);
  _Float16* FC2_16 = (_Float16*)arena(4096 * 2);
  _Float16* KW_16  = (_Float16*)arena(4096 * 2);
  _Float16* VW_16  = (_Float16*)arena(4096 * 2);
  const size_t FEAT = (size_t)BB * HWP * DD;               // 6,422,528 floats
  float* E0   = (float*)arena(FEAT * 4);
  float* E1   = (float*)arena(FEAT * 4);                   // reused as attn7 buffer
  float* KBUF = (float*)arena(FEAT * 4);
  float* VBUF = (float*)arena(FEAT * 4);
  float* STATS = (float*)arena(16 * 4);
  float* SSUM  = (float*)arena(64 * 4);
  float* SBUF  = (float*)arena(BN * 64 * 4);
  float* QBUF  = (float*)arena(BN * 64 * 4);
  float* UPD   = (float*)arena(BN * 64 * 4);
  float* GIbuf = (float*)arena(BN * 192 * 4);
  float* GHbuf = (float*)arena(BN * 192 * 4);
  float* T1    = (float*)arena(BN * 64 * 4);
  float* T2    = (float*)arena(BN * 64 * 4);
  float* SLOTS = (float*)arena(BN * 64 * 4);
  _Float16* DEC0 = (_Float16*)arena((size_t)BN * 128 * 128 * 64 * 2);
  _Float16* DEC1 = (_Float16*)arena((size_t)BN * 128 * 128 * 64 * 2);
  float* ATTN7 = E1;  // 100352*8 floats fits in E1

  auto cdiv = [](size_t a, size_t b) { return (a + b - 1) / b; };

  // ---- weight conversion (into WMMA B-fragment layout) ----
  wcvt_conv<<<cdiv(102400, 256), 256, 0, stream>>>(enc_w2, W2_16);
  wcvt_conv<<<cdiv(102400, 256), 256, 0, stream>>>(enc_w3, W3_16);
  wcvt_conv<<<cdiv(102400, 256), 256, 0, stream>>>(enc_w4, W4_16);
  wcvt_conv<<<cdiv(102400, 256), 256, 0, stream>>>(dec_w5, W5_16);
  wcvt_lin<<<16, 256, 0, stream>>>(fc1_w, FC1_16);
  wcvt_lin<<<16, 256, 0, stream>>>(fc2_w, FC2_16);
  wcvt_lin<<<16, 256, 0, stream>>>(k_w, KW_16);
  wcvt_lin<<<16, 256, 0, stream>>>(v_w, VW_16);

  // ---- encoder ----
  conv1_direct<<<cdiv(FEAT, 256), 256, 0, stream>>>(image, enc_w1, enc_b1, E0);
  {
    dim3 g(WW / 16, HH / 2, BB);
    conv5x5_64_wmma<float, float, true><<<g, 32, 0, stream>>>(E0, W2_16, enc_b2, E1, BB, HH, WW);
    conv5x5_64_wmma<float, float, true><<<g, 32, 0, stream>>>(E1, W3_16, enc_b3, E0, BB, HH, WW);
    conv5x5_64_wmma<float, float, true><<<g, 32, 0, stream>>>(E0, W4_16, enc_b4, E1, BB, HH, WW);
  }
  pos_add_enc<<<cdiv(FEAT, 256), 256, 0, stream>>>(E1, enc_pw, enc_pb);
  zero_f<<<1, 256, 0, stream>>>(STATS, 16);
  reduce_stats<<<dim3(784, BB), 256, 0, stream>>>(E1, STATS);
  apply_gln<<<cdiv(FEAT, 256), 256, 0, stream>>>(E1, STATS);
  linear64_wmma<true ><<<ROWS / 16, 32, 0, stream>>>(E1, FC1_16, fc1_b, E0, ROWS);
  linear64_wmma<false><<<ROWS / 16, 32, 0, stream>>>(E0, FC2_16, fc2_b, E1, ROWS);
  ln_rows<<<ROWS, 64, 0, stream>>>(E1, ni_g, ni_b, E0, ROWS);     // inputs -> E0
  linear64_wmma<false><<<ROWS / 16, 32, 0, stream>>>(E0, KW_16, k_b, KBUF, ROWS);
  linear64_wmma<false><<<ROWS / 16, 32, 0, stream>>>(E0, VW_16, v_b, VBUF, ROWS);

  // ---- slot attention (3 iterations) ----
  copy_f<<<cdiv(BN * 64, 256), 256, 0, stream>>>(init_sl, SLOTS, BN * 64);
  for (int it = 0; it < 3; ++it) {
    ln_rows<<<BN, 64, 0, stream>>>(SLOTS, ns_g, ns_b, SBUF, BN);
    linear_small<<<cdiv(BN * 64, 256), 256, 0, stream>>>(SBUF, q_w, q_b, QBUF, BN, 64, 0);
    zero_f<<<1, 256, 0, stream>>>(SSUM, BN);
    zero_f<<<cdiv(BN * 64, 256), 256, 0, stream>>>(UPD, BN * 64);
    attn_dots<<<dim3(HWP / 256, BB), 256, 0, stream>>>(QBUF, KBUF, ATTN7, SSUM);
    attn_updates<<<dim3(13, NSL, BB), 256, 0, stream>>>(ATTN7, VBUF, UPD);
    div_upd<<<cdiv(BN * 64, 256), 256, 0, stream>>>(UPD, SSUM);
    linear_small<<<cdiv(BN * 192, 256), 256, 0, stream>>>(UPD, gru_wih, gru_bih, GIbuf, BN, 192, 0);
    linear_small<<<cdiv(BN * 192, 256), 256, 0, stream>>>(SLOTS, gru_whh, gru_bhh, GHbuf, BN, 192, 0);
    gru_combine<<<cdiv(BN * 64, 256), 256, 0, stream>>>(GIbuf, GHbuf, SLOTS);
    ln_rows<<<BN, 64, 0, stream>>>(SLOTS, npf_g, npf_b, T1, BN);
    linear_small<<<cdiv(BN * 64, 256), 256, 0, stream>>>(T1, s1_w, s1_b, T2, BN, 64, 1);
    linear_small<<<cdiv(BN * 64, 256), 256, 0, stream>>>(T2, s2_w, s2_b, T1, BN, 64, 0);
    add_f<<<cdiv(BN * 64, 256), 256, 0, stream>>>(SLOTS, T1, BN * 64);
  }
  copy_f<<<cdiv(BN * 64, 256), 256, 0, stream>>>(SLOTS, out_slots, BN * 64);

  // ---- decoder ----
  build_decin<<<cdiv((size_t)BN * 64 * 64, 256), 256, 0, stream>>>(SLOTS, dec_pw, dec_pb, DEC0);
  deconv5x5<<<cdiv((size_t)BN * 16 * 16 * 64, 256), 256, 0, stream>>>(DEC0, dec_w1, dec_b1, DEC1, BN, 8, 8);
  deconv5x5<<<cdiv((size_t)BN * 32 * 32 * 64, 256), 256, 0, stream>>>(DEC1, dec_w2, dec_b2, DEC0, BN, 16, 16);
  deconv5x5<<<cdiv((size_t)BN * 64 * 64 * 64, 256), 256, 0, stream>>>(DEC0, dec_w3, dec_b3, DEC1, BN, 32, 32);
  deconv5x5<<<cdiv((size_t)BN * 128 * 128 * 64, 256), 256, 0, stream>>>(DEC1, dec_w4, dec_b4, DEC0, BN, 64, 64);
  {
    dim3 g(128 / 16, 128 / 2, BN);
    conv5x5_64_wmma<_Float16, _Float16, true><<<g, 32, 0, stream>>>(DEC0, W5_16, dec_b5, DEC1, BN, 128, 128);
  }
  conv6_out<<<cdiv((size_t)BN * HWP * 4, 256), 256, 0, stream>>>(DEC1, dec_w6, dec_b6, out_recons, out_masks);
}
